// BipartiteRGAT_1056561955276
// MI455X (gfx1250) — matrix operand order repaired and checked by reading, not compiled, not run on hardware
//
#include <hip/hip_runtime.h>
#include <hip/hip_bf16.h>

// ---------------- model constants ----------------
#define N_PER   10000
#define NTOT    30000
#define NEDGE   500000
#define NREL    6
#define FDIM    128
#define LABELS  5
#define SLOPE   0.2f

typedef __attribute__((ext_vector_type(16))) __bf16 v16bf;
typedef __attribute__((ext_vector_type(8)))  float  v8f;

__device__ __forceinline__ float elu_f(float x) {
    return x > 0.f ? x : (__expf(x) - 1.f);
}

// monotonic float<->uint mapping for atomicMax-based segment max
__device__ __forceinline__ unsigned fflip(float f) {
    unsigned u = __float_as_uint(f);
    return (u & 0x80000000u) ? ~u : (u | 0x80000000u);
}
__device__ __forceinline__ float funflip(unsigned u) {
    u = (u & 0x80000000u) ? (u & 0x7FFFFFFFu) : ~u;
    return __uint_as_float(u);
}

// ---------------------------------------------------------------------------
// Weight pre-pass: W [nmat][K][128] f32  ->  Wt [nmat][128][K] bf16 (transposed)
// ---------------------------------------------------------------------------
__global__ void convert_wt_kernel(const float* __restrict__ W,
                                  __bf16* __restrict__ Wt,
                                  int K, int nmat) {
    long long i = (long long)blockIdx.x * blockDim.x + threadIdx.x;
    long long total = (long long)nmat * K * FDIM;
    if (i >= total) return;
    int r   = (int)(i / ((long long)K * FDIM));
    int rem = (int)(i % ((long long)K * FDIM));
    int k = rem >> 7;
    int n = rem & (FDIM - 1);
    Wt[((size_t)r * FDIM + n) * K + k] = (__bf16)W[i];
}

// ---------------------------------------------------------------------------
// WMMA GEMM: C[M,128] = act(A[M,K] @ B[K,128] + bias)
// B passed pre-converted/transposed: Bt [128][K] bf16.
// Block = 256 threads (8 waves). Block tile 128(M) x 128(N).
// Each wave: 16 rows x 128 cols = 8 accumulators of 16x16.
// K split into full 128-chunks (branchless, fully unrolled substeps) plus one
// zero-padded tail chunk with clamped A addressing (B zero => product zero).
// Requires K % 4 == 0 (all call sites: 2000/1500/1000/128).
// ACT: 0 = none, 1 = ELU.
// ---------------------------------------------------------------------------
template<int ACT>
__launch_bounds__(256)
__global__ void gemm_bias_kernel(const float* __restrict__ A,
                                 const __bf16* __restrict__ Bt,
                                 const float* __restrict__ bias,
                                 float* __restrict__ C,
                                 int M, int K) {
    __shared__ __bf16 Bs[FDIM * 128];      // [n][kchunk], 32 KB

    const int lane  = threadIdx.x & 31;
    const int wave  = threadIdx.x >> 5;
    const int m0    = blockIdx.x * 128 + wave * 16;
    const int mrow  = m0 + (lane & 15);    // logical A row for this lane
    const int khalf = lane >> 4;           // 0: lanes 0-15, 1: lanes 16-31
    const int colb  = lane & 15;

    // branchless A row: clamp; out-of-range rows compute garbage, masked at store
    const int arow = (mrow < M) ? mrow : (M - 1);
    const float* __restrict__ ap = A + (size_t)arow * K;

    v8f acc[8];
#pragma unroll
    for (int t = 0; t < 8; ++t)
#pragma unroll
        for (int i = 0; i < 8; ++i) acc[t][i] = 0.f;

    const int kfull = K & ~127;

    // ================= full 128-K chunks: branchless =================
    for (int kb = 0; kb < kfull; kb += 128) {
        __syncthreads();
        // stage Bt chunk [128 n][128 k] into LDS as b64 copies (no guards)
        for (int j = threadIdx.x; j < FDIM * 32; j += 256) {   // 32 uint2 per row
            int n = j >> 5;
            int c = j & 31;
            *(uint2*)(&Bs[n * 128 + c * 4]) =
                *(const uint2*)(Bt + (size_t)n * K + kb + c * 4);
        }
        // speculative prefetch of next weight chunk (global_prefetch_b8)
        if (kb + 128 < K) {
            const __bf16* pf = Bt + (size_t)(threadIdx.x >> 1) * K + (kb + 128)
                             + (threadIdx.x & 1) * 64;
            __builtin_prefetch((const void*)pf, 0, 0);
        }
        __syncthreads();

#pragma unroll
        for (int kc = 0; kc < 128; kc += 32) {
            // A fragment (contiguous k order: k = 16*khalf + e), f32 -> bf16, unguarded
            v16bf afrag;
            const int klo = kb + kc + khalf * 16;
#pragma unroll
            for (int h = 0; h < 4; ++h) {
                float2 p = *(const float2*)(ap + klo + 2 * h);
                afrag[2 * h]     = (__bf16)p.x;
                afrag[2 * h + 1] = (__bf16)p.y;
            }
#pragma unroll
            for (int h = 0; h < 4; ++h) {
                float2 p = *(const float2*)(ap + klo + 8 + 2 * h);
                afrag[8 + 2 * h]     = (__bf16)p.x;
                afrag[8 + 2 * h + 1] = (__bf16)p.y;
            }
            // 8 WMMAs across N=128, B-fragments preloaded in groups of 4
#pragma unroll
            for (int g = 0; g < 2; ++g) {
                v16bf bfr[4];
#pragma unroll
                for (int t = 0; t < 4; ++t) {
                    const __bf16* bp =
                        &Bs[((g * 4 + t) * 16 + colb) * 128 + kc + khalf * 16];
#pragma unroll
                    for (int e = 0; e < 16; ++e) bfr[t][e] = bp[e];
                }
#pragma unroll
                for (int t = 0; t < 4; ++t)
                    acc[g * 4 + t] = __builtin_amdgcn_wmma_f32_16x16x32_bf16(
                        false, afrag, false, bfr[t], (short)0,
                        acc[g * 4 + t], false, false);
            }
        }
    }

    // ================= tail chunk (K % 128 != 0): zero-padded B =================
    if (kfull < K) {
        __syncthreads();
        for (int j = threadIdx.x; j < FDIM * 32; j += 256) {
            int n  = j >> 5;
            int c  = j & 31;
            int k0 = kfull + c * 4;
            uint2 v;
            if (k0 + 4 <= K) v = *(const uint2*)(Bt + (size_t)n * K + k0);
            else             v = uint2{0u, 0u};   // K%4==0 -> never partial
            *(uint2*)(&Bs[n * 128 + c * 4]) = v;
        }
        __syncthreads();

#pragma unroll
        for (int kc = 0; kc < 128; kc += 32) {
            // clamped A addressing: for k >= K the B column is zero, so the
            // (duplicated) A value contributes nothing.
            v16bf afrag;
            const int klo = kfull + kc + khalf * 16;
#pragma unroll
            for (int e = 0; e < 16; ++e) {
                int k = klo + e;
                k = (k < K) ? k : (K - 1);
                afrag[e] = (__bf16)ap[k];
            }
#pragma unroll
            for (int g = 0; g < 2; ++g) {
                v16bf bfr[4];
#pragma unroll
                for (int t = 0; t < 4; ++t) {
                    const __bf16* bp =
                        &Bs[((g * 4 + t) * 16 + colb) * 128 + kc + khalf * 16];
#pragma unroll
                    for (int e = 0; e < 16; ++e) bfr[t][e] = bp[e];
                }
#pragma unroll
                for (int t = 0; t < 4; ++t)
                    acc[g * 4 + t] = __builtin_amdgcn_wmma_f32_16x16x32_bf16(
                        false, afrag, false, bfr[t], (short)0,
                        acc[g * 4 + t], false, false);
            }
        }
    }

    // ---- epilogue: C/D layout -> lane holds col=colb, VGPR v -> row v+8*khalf
#pragma unroll
    for (int t = 0; t < 8; ++t) {
        int n = t * 16 + colb;
        float bv = bias ? bias[n] : 0.f;
#pragma unroll
        for (int v = 0; v < 8; ++v) {
            int m = m0 + khalf * 8 + v;
            if (m < M) {
                float val = acc[t][v] + bv;
                if (ACT == 1) val = elu_f(val);
                C[(size_t)m * FDIM + n] = val;
            }
        }
    }
}

// ---------------------------------------------------------------------------
// aq[i] = xw[i,:].q ; ak[i] = xw[i,:].k  over i in [0, R*NTOT)
// ---------------------------------------------------------------------------
__global__ void qk_kernel(const float* __restrict__ xw,
                          const float* __restrict__ q,
                          const float* __restrict__ k,
                          float* __restrict__ aq, float* __restrict__ ak,
                          int total) {
    int i = blockIdx.x * blockDim.x + threadIdx.x;
    if (i >= total) return;
    const float* xp = xw + (size_t)i * FDIM;
    float sq = 0.f, sk = 0.f;
#pragma unroll 4
    for (int f = 0; f < FDIM; ++f) {
        float v = xp[f];
        sq += v * q[f];
        sk += v * k[f];
    }
    aq[i] = sq;
    ak[i] = sk;
}

// alpha_e = leaky_relu(aq[etype,dst] + ak[etype,src]); segment max via flipped atomicMax
__global__ void edge_alpha_kernel(const float* __restrict__ aq,
                                  const float* __restrict__ ak,
                                  const int* __restrict__ esrc,
                                  const int* __restrict__ edst,
                                  const int* __restrict__ etyp,
                                  float* __restrict__ alpha,
                                  unsigned* __restrict__ mflip) {
    int e = blockIdx.x * blockDim.x + threadIdx.x;
    if (e >= NEDGE) return;
    int s = esrc[e], d = edst[e], r = etyp[e];
    float a = aq[r * NTOT + d] + ak[r * NTOT + s];
    a = a > 0.f ? a : SLOPE * a;
    alpha[e] = a;
    atomicMax(&mflip[d], fflip(a));
}

// a_e = exp(alpha - m[dst]); segment sum + degree
__global__ void edge_expsum_kernel(float* __restrict__ alpha,
                                   const int* __restrict__ edst,
                                   const unsigned* __restrict__ mflip,
                                   float* __restrict__ ssum,
                                   float* __restrict__ deg) {
    int e = blockIdx.x * blockDim.x + threadIdx.x;
    if (e >= NEDGE) return;
    int d = edst[e];
    float a = __expf(alpha[e] - funflip(mflip[d]));
    alpha[e] = a;                          // in-place: alpha -> a
    atomicAdd(&ssum[d], a);
    atomicAdd(&deg[d], 1.0f);
}

// coef_e = a_e / s[dst] * deg[dst]   ("f-scaled")
__global__ void edge_coef_kernel(float* __restrict__ alpha,
                                 const int* __restrict__ edst,
                                 const float* __restrict__ ssum,
                                 const float* __restrict__ deg) {
    int e = blockIdx.x * blockDim.x + threadIdx.x;
    if (e >= NEDGE) return;
    int d = edst[e];
    alpha[e] = alpha[e] / ssum[d] * deg[d];   // in-place: a -> coef
}

// agg[dst,f] += coef_e * xw[etype,src,f]  (one thread per (edge,feature))
__global__ void edge_scatter_kernel(const float* __restrict__ coef,
                                    const int* __restrict__ esrc,
                                    const int* __restrict__ edst,
                                    const int* __restrict__ etyp,
                                    const float* __restrict__ xw,
                                    float* __restrict__ agg) {
    long long t = (long long)blockIdx.x * blockDim.x + threadIdx.x;
    if (t >= (long long)NEDGE * FDIM) return;
    int e = (int)(t >> 7);
    int f = (int)(t & (FDIM - 1));
    int s = esrc[e], d = edst[e], r = etyp[e];
    float v = coef[e] * xw[((size_t)r * NTOT + s) * FDIM + f];
    atomicAdd(&agg[(size_t)d * FDIM + f], v);
}

// h = elu(agg + bias + sl)
__global__ void finalize_kernel(const float* __restrict__ agg,
                                const float* __restrict__ bias,
                                const float* __restrict__ sl,
                                float* __restrict__ h, int total) {
    int i = blockIdx.x * blockDim.x + threadIdx.x;
    if (i >= total) return;
    int f = i & (FDIM - 1);
    h[i] = elu_f(agg[i] + bias[f] + sl[i]);
}

// out[n,l] = z[n,:] @ lin2W[:,l] + lin2b[l]
__global__ void lin2_kernel(const float* __restrict__ z,
                            const float* __restrict__ W,
                            const float* __restrict__ b,
                            float* __restrict__ out) {
    int i = blockIdx.x * blockDim.x + threadIdx.x;
    if (i >= N_PER * LABELS) return;
    int n = i / LABELS, l = i % LABELS;
    const float* zp = z + (size_t)n * FDIM;
    float s = b[l];
#pragma unroll 4
    for (int f = 0; f < FDIM; ++f) s += zp[f] * W[f * LABELS + l];
    out[i] = s;
}

// ---------------------------------------------------------------------------
extern "C" void kernel_launch(void* const* d_in, const int* in_sizes, int n_in,
                              void* d_out, int out_size, void* d_ws, size_t ws_size,
                              hipStream_t stream) {
    (void)in_sizes; (void)n_in; (void)out_size; (void)ws_size;

    const float* x0     = (const float*)d_in[0];
    const float* x1     = (const float*)d_in[1];
    const float* x2     = (const float*)d_in[2];
    const float* projW0 = (const float*)d_in[3];
    const float* projb0 = (const float*)d_in[4];
    const float* projW1 = (const float*)d_in[5];
    const float* projb1 = (const float*)d_in[6];
    const float* projW2 = (const float*)d_in[7];
    const float* projb2 = (const float*)d_in[8];
    const float* slW0   = (const float*)d_in[9];
    const float* slb0   = (const float*)d_in[10];
    const float* slW1   = (const float*)d_in[11];
    const float* slb1   = (const float*)d_in[12];
    const float* slW2   = (const float*)d_in[13];
    const float* slb2   = (const float*)d_in[14];
    const float* W1     = (const float*)d_in[15];
    const float* q1     = (const float*)d_in[16];
    const float* k1     = (const float*)d_in[17];
    const float* b1     = (const float*)d_in[18];
    const float* W2     = (const float*)d_in[19];
    const float* q2     = (const float*)d_in[20];
    const float* k2     = (const float*)d_in[21];
    const float* b2     = (const float*)d_in[22];
    const float* lin1W  = (const float*)d_in[23];
    const float* lin1b  = (const float*)d_in[24];
    const float* lin2W  = (const float*)d_in[25];
    const float* lin2b  = (const float*)d_in[26];
    const int*   esrc   = (const int*)d_in[27];
    const int*   edst   = (const int*)d_in[28];
    const int*   etyp   = (const int*)d_in[29];

    // ---- workspace partition ----
    char* w = (char*)d_ws;
    auto takeb = [&](size_t nbytes) -> char* {
        char* p = w;
        w += ((nbytes + 255) & ~size_t(255));
        return p;
    };
    auto take  = [&](size_t nfloats) -> float*  { return (float*)takeb(nfloats * 4); };
    auto takeh = [&](size_t nhalf)   -> __bf16* { return (__bf16*)takeb(nhalf * 2); };

    float*    xw    = take((size_t)NREL * NTOT * FDIM);  // 92.2 MB, fits global L2
    float*    xall  = take((size_t)NTOT * FDIM);         // projected x, later h
    float*    sl    = take((size_t)NTOT * FDIM);
    float*    agg   = take((size_t)NTOT * FDIM);
    float*    aq    = take((size_t)NREL * NTOT);
    float*    ak    = take((size_t)NREL * NTOT);
    unsigned* mflip = (unsigned*)take(NTOT);
    float*    ssum  = take(NTOT);
    float*    deg   = take(NTOT);
    float*    ebuf  = take(NEDGE);                        // alpha -> a -> coef
    float*    z     = take((size_t)N_PER * FDIM);
    // pre-converted bf16 transposed weights [nmat][128][K]
    __bf16* projWt0 = takeh((size_t)2000 * FDIM);
    __bf16* projWt1 = takeh((size_t)1500 * FDIM);
    __bf16* projWt2 = takeh((size_t)1000 * FDIM);
    __bf16* slWt0   = takeh((size_t)FDIM * FDIM);
    __bf16* slWt1   = takeh((size_t)FDIM * FDIM);
    __bf16* slWt2   = takeh((size_t)FDIM * FDIM);
    __bf16* W1t     = takeh((size_t)NREL * FDIM * FDIM);
    __bf16* W2t     = takeh((size_t)NREL * FDIM * FDIM);
    __bf16* lin1Wt  = takeh((size_t)FDIM * FDIM);

    // ---- 0) convert + transpose all weights to bf16 once per call ----
    auto conv = [&](const float* W, __bf16* Wt, int K, int nmat) {
        long long tot = (long long)nmat * K * FDIM;
        convert_wt_kernel<<<(unsigned)((tot + 255) / 256), 256, 0, stream>>>(W, Wt, K, nmat);
    };
    conv(projW0, projWt0, 2000, 1);
    conv(projW1, projWt1, 1500, 1);
    conv(projW2, projWt2, 1000, 1);
    conv(slW0,   slWt0,   FDIM, 1);
    conv(slW1,   slWt1,   FDIM, 1);
    conv(slW2,   slWt2,   FDIM, 1);
    conv(W1,     W1t,     FDIM, NREL);
    conv(W2,     W2t,     FDIM, NREL);
    conv(lin1W,  lin1Wt,  FDIM, 1);

    const int gemmBlkPer = (N_PER + 127) / 128;           // 79
    const int gemmBlkTot = (NTOT + 127) / 128;            // 235

    // ---- 1) per-omic projections (ELU) -> xall ----
    gemm_bias_kernel<1><<<gemmBlkPer, 256, 0, stream>>>(x0, projWt0, projb0, xall,                      N_PER, 2000);
    gemm_bias_kernel<1><<<gemmBlkPer, 256, 0, stream>>>(x1, projWt1, projb1, xall + (size_t)N_PER*FDIM, N_PER, 1500);
    gemm_bias_kernel<1><<<gemmBlkPer, 256, 0, stream>>>(x2, projWt2, projb2, xall + (size_t)2*N_PER*FDIM, N_PER, 1000);

    // ---- 2) self-loop linears (no act) -> sl ----
    gemm_bias_kernel<0><<<gemmBlkPer, 256, 0, stream>>>(xall,                        slWt0, slb0, sl,                        N_PER, FDIM);
    gemm_bias_kernel<0><<<gemmBlkPer, 256, 0, stream>>>(xall + (size_t)N_PER*FDIM,   slWt1, slb1, sl + (size_t)N_PER*FDIM,   N_PER, FDIM);
    gemm_bias_kernel<0><<<gemmBlkPer, 256, 0, stream>>>(xall + (size_t)2*N_PER*FDIM, slWt2, slb2, sl + (size_t)2*N_PER*FDIM, N_PER, FDIM);

    // ---- RGAT layer (shared sequence) ----
    auto rgat = [&](const __bf16* Wt, const float* q, const float* k, const float* b) {
        for (int r = 0; r < NREL; ++r)
            gemm_bias_kernel<0><<<gemmBlkTot, 256, 0, stream>>>(
                xall, Wt + (size_t)r * FDIM * FDIM, nullptr,
                xw + (size_t)r * NTOT * FDIM, NTOT, FDIM);
        qk_kernel<<<(NREL * NTOT + 255) / 256, 256, 0, stream>>>(xw, q, k, aq, ak, NREL * NTOT);
        (void)hipMemsetAsync(mflip, 0, NTOT * sizeof(unsigned), stream);
        (void)hipMemsetAsync(ssum,  0, NTOT * sizeof(float), stream);
        (void)hipMemsetAsync(deg,   0, NTOT * sizeof(float), stream);
        (void)hipMemsetAsync(agg,   0, (size_t)NTOT * FDIM * sizeof(float), stream);
        const int eblk = (NEDGE + 255) / 256;
        edge_alpha_kernel <<<eblk, 256, 0, stream>>>(aq, ak, esrc, edst, etyp, ebuf, mflip);
        edge_expsum_kernel<<<eblk, 256, 0, stream>>>(ebuf, edst, mflip, ssum, deg);
        edge_coef_kernel  <<<eblk, 256, 0, stream>>>(ebuf, edst, ssum, deg);
        const long long nsc = (long long)NEDGE * FDIM;
        edge_scatter_kernel<<<(unsigned)((nsc + 255) / 256), 256, 0, stream>>>(
            ebuf, esrc, edst, etyp, xw, agg);
        finalize_kernel<<<(NTOT * FDIM + 255) / 256, 256, 0, stream>>>(
            agg, b, sl, xall, NTOT * FDIM);   // xall <- h (safe: last read of xall done)
    };

    rgat(W1t, q1, k1, b1);   // h1 in xall
    rgat(W2t, q2, k2, b2);   // h2 in xall

    // ---- head ----
    gemm_bias_kernel<1><<<gemmBlkPer, 256, 0, stream>>>(xall, lin1Wt, lin1b, z, N_PER, FDIM);
    lin2_kernel<<<(N_PER * LABELS + 255) / 256, 256, 0, stream>>>(z, lin2W, lin2b, (float*)d_out);
}
